// SelfAttentionModule_8160437862677
// MI455X (gfx1250) — matrix operand order repaired and checked by reading, not compiled, not run on hardware
//
#include <hip/hip_runtime.h>

typedef __attribute__((ext_vector_type(16))) _Float16 v16h;
typedef __attribute__((ext_vector_type(8)))  _Float16 v8h;
typedef __attribute__((ext_vector_type(8)))  float    v8f;
typedef __attribute__((ext_vector_type(4)))  int      v4i;

#define WMMA_F16(a, b, c) \
    __builtin_amdgcn_wmma_f32_16x16x32_f16(false, (a), false, (b), (short)0, (c), false, false)

#if __has_builtin(__builtin_amdgcn_global_load_async_to_lds_b128) && \
    __has_builtin(__builtin_amdgcn_s_wait_asynccnt)
#define USE_ASYNC_LDS 1
#else
#define USE_ASYNC_LDS 0
#endif

constexpr int  Bsz = 4;
constexpr int  S   = 4096;
constexpr int  D   = 1024;
constexpr long BS  = (long)Bsz * S;   // 16384 rows total
constexpr int  QR  = 32;              // query rows per attention block

// 16 contiguous f16 (B fragment: K-major slice, 32 bytes) -> two b128 loads
__device__ __forceinline__ v16h ld16c(const _Float16* p) {
    v8h lo = *(const v8h*)p;
    v8h hi = *(const v8h*)(p + 8);
    return __builtin_shufflevector(lo, hi, 0,1,2,3,4,5,6,7,8,9,10,11,12,13,14,15);
}
// A fragment row slice: halves [0..7] and [16..23] relative to rowp
// (rowp = row_base + kk + 8*sel per the ISA 16-bit A layout)
__device__ __forceinline__ v16h ldAf(const _Float16* rowp) {
    v8h lo = *(const v8h*)rowp;
    v8h hi = *(const v8h*)(rowp + 16);
    return __builtin_shufflevector(lo, hi, 0,1,2,3,4,5,6,7,8,9,10,11,12,13,14,15);
}

// ---------------------------------------------------------------------------
// Kernel 1: fused QKV projection (q = x * W^T + b), f32 in -> f16 out.
// Block = 64 rows x 64 cols, 4 waves; each wave: 16 cols, 4 row-tiles of 16.
// Each weight B-fragment feeds 4 WMMAs (one per row-tile); weight fragments
// are double-buffered across the K loop so WMMAs overlap the next loads.
// Q, K row-major [BS][D]; V transposed [D][BS] for contiguous PV B-fragments.
// ---------------------------------------------------------------------------
__global__ void __launch_bounds__(128)
qkv_proj_kernel(const float* __restrict__ x,
                const float* __restrict__ Wq, const float* __restrict__ bq,
                const float* __restrict__ Wk, const float* __restrict__ bk,
                const float* __restrict__ Wv, const float* __restrict__ bv,
                _Float16* __restrict__ Qh, _Float16* __restrict__ Kh,
                _Float16* __restrict__ Vt)
{
    const int lane = threadIdx.x & 31;
    const int wave = threadIdx.x >> 5;           // 0..3
    const long rowbase = (long)blockIdx.y * 64;  // 4 row-tiles over B*S
    const int  colbase = blockIdx.x * 64 + wave * 16;

    const int nn  = lane & 15;
    const int sel = lane >> 4;

    v8f cq[4] = {}, ck[4] = {}, cv[4] = {};

    const long wrow = (long)(colbase + nn) * D;

    auto loadW3 = [&](int d0, v16h& fq, v16h& fk, v16h& fv) {
#pragma unroll
        for (int j = 0; j < 8; ++j) {
            const int kbB = sel * 16 + 2 * j;
            const float2 wq2 = *(const float2*)(Wq + wrow + d0 + kbB);
            const float2 wk2 = *(const float2*)(Wk + wrow + d0 + kbB);
            const float2 wv2 = *(const float2*)(Wv + wrow + d0 + kbB);
            fq[2 * j] = (_Float16)wq2.x; fq[2 * j + 1] = (_Float16)wq2.y;
            fk[2 * j] = (_Float16)wk2.x; fk[2 * j + 1] = (_Float16)wk2.y;
            fv[2 * j] = (_Float16)wv2.x; fv[2 * j + 1] = (_Float16)wv2.y;
        }
    };
    auto loadA = [&](int rt, int d0) -> v16h {
        const float* xr = x + (rowbase + 16 * rt + nn) * (long)D + d0;
        v16h a;
#pragma unroll
        for (int j = 0; j < 8; ++j) {
            const int kbA = sel * 16 + 2 * j;
            const float2 xv = *(const float2*)(xr + kbA);
            a[2 * j]     = (_Float16)xv.x;
            a[2 * j + 1] = (_Float16)xv.y;
        }
        return a;
    };

    v16h fq, fk, fv;
    loadW3(0, fq, fk, fv);
    for (int d0 = 0; d0 < D; d0 += 32) {
        const int dn = (d0 + 32 < D) ? d0 + 32 : d0;   // rolling prefetch
        v16h nq, nk, nv;
        loadW3(dn, nq, nk, nv);
        const v16h a0 = loadA(0, d0);
        const v16h a1 = loadA(1, d0);
        const v16h a2 = loadA(2, d0);
        const v16h a3 = loadA(3, d0);
        cq[0] = WMMA_F16(a0, fq, cq[0]); ck[0] = WMMA_F16(a0, fk, ck[0]); cv[0] = WMMA_F16(a0, fv, cv[0]);
        cq[1] = WMMA_F16(a1, fq, cq[1]); ck[1] = WMMA_F16(a1, fk, ck[1]); cv[1] = WMMA_F16(a1, fv, cv[1]);
        cq[2] = WMMA_F16(a2, fq, cq[2]); ck[2] = WMMA_F16(a2, fk, ck[2]); cv[2] = WMMA_F16(a2, fv, cv[2]);
        cq[3] = WMMA_F16(a3, fq, cq[3]); ck[3] = WMMA_F16(a3, fk, ck[3]); cv[3] = WMMA_F16(a3, fv, cv[3]);
        fq = nq; fk = nk; fv = nv;
    }

    const float biasq = bq[colbase + nn];
    const float biask = bk[colbase + nn];
    const float biasv = bv[colbase + nn];

#pragma unroll
    for (int rt = 0; rt < 4; ++rt) {
#pragma unroll
        for (int j = 0; j < 8; ++j) {
            const long gr = rowbase + 16 * rt + j + 8 * sel;
            Qh[gr * D + colbase + nn]          = (_Float16)(cq[rt][j] + biasq);
            Kh[gr * D + colbase + nn]          = (_Float16)(ck[rt][j] + biask);
            Vt[(long)(colbase + nn) * BS + gr] = (_Float16)(cv[rt][j] + biasv);
        }
    }
}

// ---------------------------------------------------------------------------
// Kernel 2: flash attention. One block = (batch, 32-query tile); 8 waves.
// Wave w owns output D-slice [128w,128w+128) and key sub-tile [16w,16w+16)
// of each 128-key chunk. B fragments are shared by the two 16-row tiles.
// ---------------------------------------------------------------------------
__global__ void __launch_bounds__(256)
attention_kernel(const _Float16* __restrict__ Qh,
                 const _Float16* __restrict__ Kh,
                 const _Float16* __restrict__ Vt,
                 float* __restrict__ out)
{
    __shared__ __attribute__((aligned(16))) _Float16 qtile[QR * D];   // 64 KB
    __shared__ float    sbuf[QR * 128];                               // 16 KB
    __shared__ __attribute__((aligned(16))) _Float16 pbuf[QR * 128];  // 8 KB
    __shared__ float    part[8 * QR];
    __shared__ float    mbuf[QR], lbuf[QR], albuf[QR];

    const int tid  = threadIdx.x;
    const int lane = tid & 31;
    const int wave = tid >> 5;                  // 0..7
    const int nn   = lane & 15;
    const int sel  = lane >> 4;

    const int  batch = blockIdx.x / (S / QR);
    const int  qt    = blockIdx.x % (S / QR);
    const long qrow  = (long)batch * S + (long)qt * QR;
    const long krow0 = (long)batch * S;
    const float scale = 0.03125f;               // 1/sqrt(1024)

    // ---- stage Q tile (QR x D f16 = 64 KB) into LDS --------------------
#if USE_ASYNC_LDS
    {
        typedef __attribute__((address_space(1))) v4i* gp_t;
        typedef __attribute__((address_space(3))) v4i* lp_t;
        gp_t src = (gp_t)(Qh + qrow * D);
        lp_t dst = (lp_t)qtile;
        for (int i = tid; i < (QR * D) / 8; i += 256) {
            __builtin_amdgcn_global_load_async_to_lds_b128(src + i, dst + i, 0, 0);
        }
        __builtin_amdgcn_s_wait_asynccnt(0);
    }
#else
    {
        const uint4* src = (const uint4*)(Qh + qrow * D);
        uint4*       dst = (uint4*)qtile;
        for (int i = tid; i < (QR * D) / 8; i += 256) dst[i] = src[i];
    }
#endif
    if (tid < QR) { mbuf[tid] = -3.0e38f; lbuf[tid] = 0.0f; }
    __syncthreads();

    v8f acc0[8] = {};   // rows 0..15  x this wave's 128 cols
    v8f acc1[8] = {};   // rows 16..31 x this wave's 128 cols

    for (int kc = 0; kc < S; kc += 128) {
        // ---- scores: two 16x16 tiles vs keys [kc+16w, +16) ---------------
        const _Float16* kptr =
            Kh + (krow0 + kc + 16 * wave + nn) * (long)D + 16 * sel;
        // prefetch next chunk's K row for this lane
        {
            const int kcn = (kc + 128 < S) ? kc + 128 : kc;
            __builtin_prefetch(Kh + (krow0 + kcn + 16 * wave + nn) * (long)D, 0, 1);
        }
        v8f sc0 = {}, sc1 = {};
        v16h bcur = ld16c(kptr);
#pragma unroll 4
        for (int kk = 0; kk < D; kk += 32) {
            const int kn = (kk + 32 < D) ? kk + 32 : kk;   // rolling prefetch
            v16h bnxt = ld16c(kptr + kn);
            v16h a0 = ld16c(qtile + nn * D        + kk + 16 * sel);
            v16h a1 = ld16c(qtile + (16 + nn) * D + kk + 16 * sel);
            sc0 = WMMA_F16(a0, bcur, sc0);
            sc1 = WMMA_F16(a1, bcur, sc1);
            bcur = bnxt;
        }
#pragma unroll
        for (int j = 0; j < 8; ++j) {
            sbuf[(j + 8 * sel) * 128      + 16 * wave + nn] = sc0[j] * scale;
            sbuf[(16 + j + 8 * sel) * 128 + 16 * wave + nn] = sc1[j] * scale;
        }
        __syncthreads();

        // ---- online softmax over the 128-key chunk (QR=32 rows) ----------
        {
            const int row = tid & 31, seg = tid >> 5;      // 8 segs x 16 cols
            float mx = -3.0e38f;
#pragma unroll
            for (int c = 0; c < 16; ++c)
                mx = fmaxf(mx, sbuf[row * 128 + seg * 16 + c]);
            part[seg * QR + row] = mx;
        }
        __syncthreads();
        if (tid < QR) {
            const float mold = mbuf[tid];
            float mx = mold;
#pragma unroll
            for (int s2 = 0; s2 < 8; ++s2) mx = fmaxf(mx, part[s2 * QR + tid]);
            mbuf[tid]  = mx;
            albuf[tid] = __expf(mold - mx);
        }
        __syncthreads();
        {
            const int row = tid & 31, seg = tid >> 5;
            const float mx = mbuf[row];
            float sum = 0.f;
#pragma unroll
            for (int c = 0; c < 16; ++c) {
                const float p = __expf(sbuf[row * 128 + seg * 16 + c] - mx);
                pbuf[row * 128 + seg * 16 + c] = (_Float16)p;
                sum += p;
            }
            part[seg * QR + row] = sum;
        }
        __syncthreads();
        if (tid < QR) {
            float l = lbuf[tid] * albuf[tid];
#pragma unroll
            for (int s2 = 0; s2 < 8; ++s2) l += part[s2 * QR + tid];
            lbuf[tid] = l;
        }
        __syncthreads();

        // ---- rescale accumulators ---------------------------------------
#pragma unroll
        for (int j = 0; j < 8; ++j) {
            const float al0 = albuf[j + 8 * sel];
            const float al1 = albuf[16 + j + 8 * sel];
#pragma unroll
            for (int t = 0; t < 8; ++t) { acc0[t][j] *= al0; acc1[t][j] *= al1; }
        }

        // ---- PV: acc += P(32x128) x V(128 x 128-slice), pipelined --------
        auto vfrag = [&](int it) -> v16h {
            const long col = 128 * wave + 16 * (it & 7) + nn;
            return ld16c(Vt + col * BS + krow0 + kc + ((it >> 3) << 5) + 16 * sel);
        };
        v16h bcur2 = vfrag(0);
        v16h a0 = {}, a1 = {};
#pragma unroll
        for (int it = 0; it < 32; ++it) {       // (kk2 = (it>>3)*32, t = it&7)
            v16h bnxt = vfrag(it < 31 ? it + 1 : 31);
            if ((it & 7) == 0) {
                const int kk2 = (it >> 3) << 5;
                a0 = ld16c(pbuf + nn * 128        + kk2 + 16 * sel);
                a1 = ld16c(pbuf + (16 + nn) * 128 + kk2 + 16 * sel);
            }
            const int t = it & 7;
            acc0[t] = WMMA_F16(a0, bcur2, acc0[t]);
            acc1[t] = WMMA_F16(a1, bcur2, acc1[t]);
            bcur2 = bnxt;
        }
        // sbuf/pbuf reuse fenced by the barriers at the top of next iter.
    }

    // ---- normalize and store -------------------------------------------
#pragma unroll
    for (int j = 0; j < 8; ++j) {
        const float inv0 = 1.0f / lbuf[j + 8 * sel];
        const float inv1 = 1.0f / lbuf[16 + j + 8 * sel];
        const long  g0   = qrow + j + 8 * sel;
        const long  g1   = qrow + 16 + j + 8 * sel;
#pragma unroll
        for (int t = 0; t < 8; ++t) {
            out[g0 * D + 128 * wave + 16 * t + nn] = acc0[t][j] * inv0;
            out[g1 * D + 128 * wave + 16 * t + nn] = acc1[t][j] * inv1;
        }
    }
}

// ---------------------------------------------------------------------------
extern "C" void kernel_launch(void* const* d_in, const int* in_sizes, int n_in,
                              void* d_out, int out_size, void* d_ws, size_t ws_size,
                              hipStream_t stream)
{
    (void)in_sizes; (void)n_in; (void)out_size; (void)ws_size;

    const float* x  = (const float*)d_in[0];
    const float* Wq = (const float*)d_in[1];
    const float* bq = (const float*)d_in[2];
    const float* Wk = (const float*)d_in[3];
    const float* bk = (const float*)d_in[4];
    const float* Wv = (const float*)d_in[5];
    const float* bv = (const float*)d_in[6];
    float* out = (float*)d_out;

    // Workspace: Q (32MB) + K (32MB) + V^T (32MB), all f16.
    _Float16* Qh = (_Float16*)d_ws;
    _Float16* Kh = Qh + BS * (long)D;
    _Float16* Vt = Kh + BS * (long)D;

    dim3 g1(D / 64, BS / 64);                    // (16, 256) blocks, 4 waves
    qkv_proj_kernel<<<g1, 128, 0, stream>>>(x, Wq, bq, Wk, bk, Wv, bv, Qh, Kh, Vt);

    attention_kernel<<<BS / QR, 256, 0, stream>>>(Qh, Kh, Vt, out);
}
